// FixedPointHGRNAttention_52544629899395
// MI455X (gfx1250) — compile-verified, matrix operand order checked by reading, not run on hardware
//
#include <hip/hip_runtime.h>
#include <cstdint>
#include <cstddef>

// ---------------- problem constants (match reference) ----------------
constexpr int Dh  = 2048;           // hidden size (= K = N)
constexpr int Bb  = 4;              // batch
constexpr int Tt  = 2048;           // seq len
constexpr int Mr  = Bb * Tt;        // 8192 rows
constexpr int CHUNK = 256;          // scan chunk
constexpr int NC    = Tt / CHUNK;   // 8 chunks per channel

typedef __attribute__((ext_vector_type(16))) _Float16 v16h;
typedef __attribute__((ext_vector_type(8)))  _Float16 v8h;
typedef __attribute__((ext_vector_type(8)))  float    v8f;

static __device__ __forceinline__ float sigmoidf_dev(float x) {
    return 1.0f / (1.0f + __expf(-x));
}

// ---------------- CDNA5 async global->LDS (ASYNCcnt-tracked) ----------------
// GVS mode: mem_addr = SGPR_base + VGPR_i32_offset + inst_offset.
// VGPR %0 holds the per-lane LDS byte address (low 32 bits of generic pointer
// == LDS offset per the aperture truncation rule).
static __device__ __forceinline__ void async_ld16(void* ldsp, uint32_t voff,
                                                  const _Float16* base) {
    asm volatile("global_load_async_to_lds_b128 %0, %1, %2"
                 :
                 : "v"((uint32_t)(uintptr_t)ldsp), "v"(voff), "s"(base)
                 : "memory");
}

// ---------------- tiny elementwise pre-passes ----------------

// x -> to_fixed(x) in f16  (round(x*256)/256 is exact in f16 for |x|<8)
__global__ void k_quant_x(const float* __restrict__ x, _Float16* __restrict__ x16, int n) {
    int i = blockIdx.x * blockDim.x + threadIdx.x;
    if (i < n) {
        float v = rintf(x[i] * 256.0f) * 0.00390625f;
        x16[i] = (_Float16)v;
    }
}

// transpose+convert three [K][N] f32 weights into [N][K] f16
__global__ void k_transpose_w3(const float* __restrict__ w0, const float* __restrict__ w1,
                               const float* __restrict__ w2,
                               _Float16* __restrict__ t0, _Float16* __restrict__ t1,
                               _Float16* __restrict__ t2) {
    const float* w = (blockIdx.z == 0) ? w0 : (blockIdx.z == 1) ? w1 : w2;
    _Float16*    t = (blockIdx.z == 0) ? t0 : (blockIdx.z == 1) ? t1 : t2;
    __shared__ float tile[32][33];
    const int n0 = blockIdx.x * 32, k0 = blockIdx.y * 32;
    const int tx = threadIdx.x, ty = threadIdx.y;   // 32 x 8
#pragma unroll
    for (int j = 0; j < 4; ++j)
        tile[ty * 4 + j][tx] = w[(size_t)(k0 + ty * 4 + j) * Dh + n0 + tx];
    __syncthreads();
#pragma unroll
    for (int j = 0; j < 4; ++j)
        t[(size_t)(n0 + ty * 4 + j) * Dh + k0 + tx] = (_Float16)tile[tx][ty * 4 + j];
}

// w_o is already [N=o][K=e]: plain convert
__global__ void k_convert_wo(const float* __restrict__ w, _Float16* __restrict__ o, int n) {
    int i = blockIdx.x * blockDim.x + threadIdx.x;
    if (i < n) o[i] = (_Float16)w[i];
}

// ---------------- WMMA fragment loader (wave32 layout) ----------------
// LDS tile region: 16 rows x 32 f16, row-contiguous (64B rows).
// lane 0-15 : row = lane,    K 0..7  in v0-3, K16..23 in v4-7
// lane 16-31: row = lane-16, K 8..15 in v0-3, K24..31 in v4-7
static __device__ __forceinline__ v16h load_frag(const _Float16* lds, int lane) {
    const int r  = lane & 15;
    const int hi = lane >> 4;
    const v8h a = *(const v8h*)(lds + r * 32 + hi * 8);
    const v8h b = *(const v8h*)(lds + r * 32 + hi * 8 + 16);
    return __builtin_shufflevector(a, b, 0, 1, 2, 3, 4, 5, 6, 7, 8, 9, 10, 11, 12, 13, 14, 15);
}

static __device__ __forceinline__ v8f wmma_f16(v16h a, v16h b, v8f c) {
    return __builtin_amdgcn_wmma_f32_16x16x32_f16(false, a, false, b, (short)0, c, false, false);
}

// ---------------- fused triple ternary GEMM (i, f, g) ----------------
// block tile = 128(M) x 64(N); 8 wave32 waves arranged 4(M) x 2(N):
// each wave owns a 32x32 tile per output -> every LDS fragment is reused
// (A frags x6, B frags x2), cutting ds traffic per WMMA to 1.33 loads.
// Double-buffered LDS fed by async global->LDS; ASYNCcnt pipelining.
__global__ __launch_bounds__(256) void k_gemm3(
    const _Float16* __restrict__ A,                 // [Mr][Dh] f16 (to_fixed(x))
    const _Float16* __restrict__ Wi, const _Float16* __restrict__ Wf,
    const _Float16* __restrict__ Wg,                // each [N][K] f16
    const float* __restrict__ si, const float* __restrict__ sf, const float* __restrict__ sg,
    float* __restrict__ Fo, float* __restrict__ Io, float* __restrict__ Go) {
    __shared__ __align__(16) _Float16 ldsA[2][128 * 32];     // 2 x 8KB
    __shared__ __align__(16) _Float16 ldsB[2][3][64 * 32];   // 2 x 12KB

    const int tid  = threadIdx.x;
    const int lane = tid & 31;
    const int wave = tid >> 5;
    const int mw   = wave & 3;      // M group: rows mw*32 .. mw*32+31
    const int nw   = wave >> 2;     // N half:  cols nw*32 .. nw*32+31
    const int n0 = blockIdx.x * 64;
    const int m0 = blockIdx.y * 128;

    // acc[matrix 0=I,1=F,2=G][n-subtile j][a-frag ai]
    v8f acc[3][2][2];
#pragma unroll
    for (int m = 0; m < 3; ++m)
#pragma unroll
        for (int j = 0; j < 2; ++j)
#pragma unroll
            for (int ai = 0; ai < 2; ++ai) acc[m][j][ai] = (v8f)0.0f;

    // loop-invariant per-thread staging coordinates (16B chunks)
    const int rowT = tid >> 2;                 // 0..63
    const int colT = (tid & 3) * 8;            // f16 units
    const uint32_t voff = (uint32_t)((rowT * Dh + colT) * sizeof(_Float16));
    const uint32_t voffLo = (uint32_t)(((rowT + 64) * Dh + colT) * sizeof(_Float16));

    // stage K-tile k into LDS buffer `buf` (5 async b128 ops per thread)
    auto stage = [&](int buf, int k) {
        const _Float16* ba = A + (size_t)m0 * Dh + k;
        async_ld16(&ldsA[buf][rowT * 32 + colT], voff, ba);
        async_ld16(&ldsA[buf][(rowT + 64) * 32 + colT], voffLo, ba);
        async_ld16(&ldsB[buf][0][rowT * 32 + colT], voff, Wi + (size_t)n0 * Dh + k);
        async_ld16(&ldsB[buf][1][rowT * 32 + colT], voff, Wf + (size_t)n0 * Dh + k);
        async_ld16(&ldsB[buf][2][rowT * 32 + colT], voff, Wg + (size_t)n0 * Dh + k);
    };

    auto compute = [&](int buf) {
        const _Float16* abase = &ldsA[buf][mw * 32 * 32];
        const v16h a0 = load_frag(abase, lane);
        const v16h a1 = load_frag(abase + 16 * 32, lane);
#pragma unroll
        for (int m = 0; m < 3; ++m) {
#pragma unroll
            for (int j = 0; j < 2; ++j) {
                v16h b = load_frag(&ldsB[buf][m][(nw * 2 + j) * 16 * 32], lane);
                acc[m][j][0] = wmma_f16(a0, b, acc[m][j][0]);
                acc[m][j][1] = wmma_f16(a1, b, acc[m][j][1]);
            }
        }
    };

    stage(0, 0);
    int cur = 0;
    for (int k0 = 0; k0 < Dh - 32; k0 += 32) {
        stage(cur ^ 1, k0 + 32);                        // prefetch next tile
        asm volatile("s_wait_asynccnt 5" ::: "memory"); // drain previous stage only
        __syncthreads();                                // buf[cur] visible to all waves
        compute(cur);
        __syncthreads();                                // reads done before restage
        cur ^= 1;
    }
    asm volatile("s_wait_asynccnt 0" ::: "memory");
    __syncthreads();
    compute(cur);                                       // final (peeled) K-tile

    // epilogue: scales + sigmoid + (1-f)*i gating
    const int r15 = lane & 15;
    const int hi  = lane >> 4;
#pragma unroll
    for (int j = 0; j < 2; ++j) {
        const int n = n0 + nw * 32 + j * 16 + r15;
        const float vsi = si[n], vsf = sf[n], vsg = sg[n];
#pragma unroll
        for (int ai = 0; ai < 2; ++ai) {
#pragma unroll
            for (int r = 0; r < 8; ++r) {
                const int m = m0 + mw * 32 + ai * 16 + r + hi * 8;
                const size_t idx = (size_t)m * Dh + n;
                const float fv = sigmoidf_dev(acc[1][j][ai][r] * vsf);
                const float iv = (1.0f - fv) * (acc[0][j][ai][r] * vsi);
                const float gv = acc[2][j][ai][r] * vsg;
                Fo[idx] = fv; Io[idx] = iv; Go[idx] = gv;
            }
        }
    }
}

// ---------------- chunked gated linear scan: h_t = f_t h_{t-1} + i_t ----------
__global__ void k_scan_chunks(const float* __restrict__ F, const float* __restrict__ I,
                              float* __restrict__ P, float* __restrict__ S) {
    const int gid = blockIdx.x * blockDim.x + threadIdx.x;   // [b][ch][d]
    if (gid >= Bb * NC * Dh) return;
    const int b  = gid / (NC * Dh);
    const int r  = gid % (NC * Dh);
    const int ch = r / Dh;
    const int d  = r % Dh;
    size_t base = ((size_t)b * Tt + (size_t)ch * CHUNK) * Dh + d;
    float p = 1.0f, s = 0.0f;
    for (int t = 0; t < CHUNK; ++t) {
        const float f = F[base];
        s = s * f + I[base];
        p *= f;
        base += Dh;
    }
    P[gid] = p; S[gid] = s;
}

__global__ void k_scan_carry(const float* __restrict__ P, const float* __restrict__ S,
                             float* __restrict__ Cin) {
    const int gid = blockIdx.x * blockDim.x + threadIdx.x;   // [b][d]
    if (gid >= Bb * Dh) return;
    const int b = gid / Dh, d = gid % Dh;
    float carry = 0.0f;
#pragma unroll
    for (int ch = 0; ch < NC; ++ch) {
        const size_t idx = ((size_t)b * NC + ch) * Dh + d;
        Cin[idx] = carry;
        carry = S[idx] + carry * P[idx];
    }
}

__global__ void k_scan_apply(const float* __restrict__ F, float* __restrict__ IH,
                             const float* __restrict__ Cin) {
    const int gid = blockIdx.x * blockDim.x + threadIdx.x;
    if (gid >= Bb * NC * Dh) return;
    const int b  = gid / (NC * Dh);
    const int r  = gid % (NC * Dh);
    const int ch = r / Dh;
    const int d  = r % Dh;
    size_t base = ((size_t)b * Tt + (size_t)ch * CHUNK) * Dh + d;
    float h = Cin[gid];
    for (int t = 0; t < CHUNK; ++t) {
        h = h * F[base] + IH[base];
        IH[base] = h;
        base += Dh;
    }
}

// ---------------- RMSNorm + swish gate + s_o fold -> f16 ----------------
__global__ __launch_bounds__(256) void k_norm_gate(
    const float* __restrict__ H, const float* __restrict__ G,
    const float* __restrict__ so, const float* __restrict__ gw,
    _Float16* __restrict__ O16) {
    __shared__ float red[256];
    const int row = blockIdx.x;
    const size_t base = (size_t)row * Dh;
    const int tid = threadIdx.x;
    float ss = 0.0f;
    for (int e = tid; e < Dh; e += 256) { const float v = H[base + e]; ss += v * v; }
    red[tid] = ss;
    __syncthreads();
    for (int s = 128; s > 0; s >>= 1) {
        if (tid < s) red[tid] += red[tid + s];
        __syncthreads();
    }
    const float rms = rsqrtf(red[0] / (float)Dh + 1e-5f);
    for (int e = tid; e < Dh; e += 256) {
        const float g = G[base + e];
        const float v = H[base + e] * rms * gw[e] * (g * sigmoidf_dev(g)) * so[e];
        O16[base + e] = (_Float16)v;
    }
}

// ---------------- output GEMM (o' @ w_o^T) with to_fixed epilogue ----------
__global__ __launch_bounds__(256) void k_gemm_out(
    const _Float16* __restrict__ A,      // [Mr][Dh] f16 (o * s_o)
    const _Float16* __restrict__ Wo,     // [N=o][K=e] f16
    float* __restrict__ out) {
    __shared__ __align__(16) _Float16 ldsA[2][128 * 32];
    __shared__ __align__(16) _Float16 ldsB[2][64 * 32];

    const int tid  = threadIdx.x;
    const int lane = tid & 31;
    const int wave = tid >> 5;
    const int mw   = wave & 3;
    const int nw   = wave >> 2;
    const int n0 = blockIdx.x * 64;
    const int m0 = blockIdx.y * 128;

    v8f acc[2][2];   // [n-subtile j][a-frag ai]
#pragma unroll
    for (int j = 0; j < 2; ++j)
#pragma unroll
        for (int ai = 0; ai < 2; ++ai) acc[j][ai] = (v8f)0.0f;

    const int rowT = tid >> 2;
    const int colT = (tid & 3) * 8;
    const uint32_t voff = (uint32_t)((rowT * Dh + colT) * sizeof(_Float16));
    const uint32_t voffLo = (uint32_t)(((rowT + 64) * Dh + colT) * sizeof(_Float16));

    auto stage = [&](int buf, int k) {
        const _Float16* ba = A + (size_t)m0 * Dh + k;
        async_ld16(&ldsA[buf][rowT * 32 + colT], voff, ba);
        async_ld16(&ldsA[buf][(rowT + 64) * 32 + colT], voffLo, ba);
        async_ld16(&ldsB[buf][rowT * 32 + colT], voff, Wo + (size_t)n0 * Dh + k);
    };

    auto compute = [&](int buf) {
        const _Float16* abase = &ldsA[buf][mw * 32 * 32];
        const v16h a0 = load_frag(abase, lane);
        const v16h a1 = load_frag(abase + 16 * 32, lane);
#pragma unroll
        for (int j = 0; j < 2; ++j) {
            v16h b = load_frag(&ldsB[buf][(nw * 2 + j) * 16 * 32], lane);
            acc[j][0] = wmma_f16(a0, b, acc[j][0]);
            acc[j][1] = wmma_f16(a1, b, acc[j][1]);
        }
    };

    stage(0, 0);
    int cur = 0;
    for (int k0 = 0; k0 < Dh - 32; k0 += 32) {
        stage(cur ^ 1, k0 + 32);
        asm volatile("s_wait_asynccnt 3" ::: "memory");
        __syncthreads();
        compute(cur);
        __syncthreads();
        cur ^= 1;
    }
    asm volatile("s_wait_asynccnt 0" ::: "memory");
    __syncthreads();
    compute(cur);

    const int r15 = lane & 15;
    const int hi  = lane >> 4;
#pragma unroll
    for (int j = 0; j < 2; ++j) {
        const int n = n0 + nw * 32 + j * 16 + r15;
#pragma unroll
        for (int ai = 0; ai < 2; ++ai) {
#pragma unroll
            for (int r = 0; r < 8; ++r) {
                const int m = m0 + mw * 32 + ai * 16 + r + hi * 8;
                out[(size_t)m * Dh + n] = rintf(acc[j][ai][r] * 256.0f) * 0.00390625f;
            }
        }
    }
}

// ---------------- host launcher ----------------
extern "C" void kernel_launch(void* const* d_in, const int* in_sizes, int n_in,
                              void* d_out, int out_size, void* d_ws, size_t ws_size,
                              hipStream_t stream) {
    (void)in_sizes; (void)n_in; (void)out_size; (void)ws_size;

    const float* x    = (const float*)d_in[0];
    const float* w_i  = (const float*)d_in[1];
    const float* w_f  = (const float*)d_in[2];
    const float* w_g  = (const float*)d_in[3];
    const float* w_o  = (const float*)d_in[4];
    const float* s_i  = (const float*)d_in[5];
    const float* s_f  = (const float*)d_in[6];
    const float* s_g  = (const float*)d_in[7];
    const float* s_o  = (const float*)d_in[8];
    const float* gnw  = (const float*)d_in[9];
    float* out = (float*)d_out;

    // workspace carve-up
    char* p = (char*)d_ws;
    _Float16* x16 = (_Float16*)p;             p += (size_t)Mr * Dh * sizeof(_Float16); // reused as o16
    _Float16* wti = (_Float16*)p;             p += (size_t)Dh * Dh * sizeof(_Float16);
    _Float16* wtf = (_Float16*)p;             p += (size_t)Dh * Dh * sizeof(_Float16);
    _Float16* wtg = (_Float16*)p;             p += (size_t)Dh * Dh * sizeof(_Float16);
    _Float16* wo16 = (_Float16*)p;            p += (size_t)Dh * Dh * sizeof(_Float16);
    float* Fb = (float*)p;                    p += (size_t)Mr * Dh * sizeof(float);
    float* Ib = (float*)p;                    p += (size_t)Mr * Dh * sizeof(float);   // becomes H in-place
    float* Gb = (float*)p;                    p += (size_t)Mr * Dh * sizeof(float);
    float* Pc = (float*)p;                    p += (size_t)Bb * NC * Dh * sizeof(float);
    float* Sc = (float*)p;                    p += (size_t)Bb * NC * Dh * sizeof(float);
    float* Ci = (float*)p;                    p += (size_t)Bb * NC * Dh * sizeof(float);
    _Float16* o16 = x16;                      // x16 dead after k_gemm3

    {
        const int n = Mr * Dh;
        k_quant_x<<<(n + 255) / 256, 256, 0, stream>>>(x, x16, n);
    }
    k_transpose_w3<<<dim3(Dh / 32, Dh / 32, 3), dim3(32, 8), 0, stream>>>(
        w_i, w_f, w_g, wti, wtf, wtg);
    {
        const int n = Dh * Dh;
        k_convert_wo<<<(n + 255) / 256, 256, 0, stream>>>(w_o, wo16, n);
    }
    k_gemm3<<<dim3(Dh / 64, Mr / 128), 256, 0, stream>>>(
        x16, wti, wtf, wtg, s_i, s_f, s_g, Fb, Ib, Gb);
    k_scan_chunks<<<(Bb * NC * Dh) / 256, 256, 0, stream>>>(Fb, Ib, Pc, Sc);
    k_scan_carry<<<(Bb * Dh) / 256, 256, 0, stream>>>(Pc, Sc, Ci);
    k_scan_apply<<<(Bb * NC * Dh) / 256, 256, 0, stream>>>(Fb, Ib, Ci);
    k_norm_gate<<<Mr, 256, 0, stream>>>(Ib, Gb, s_o, gnw, o16);
    k_gemm_out<<<dim3(Dh / 64, Mr / 128), 256, 0, stream>>>(o16, wo16, out);
}
